// OperatorBank_49082886259340
// MI455X (gfx1250) — compile-verified, matrix operand order checked by reading, not run on hardware
//
#include <hip/hip_runtime.h>
#include <hip/hip_bf16.h>

// OperatorBank: B=4096, E=1024, O=16, H=1024.
// Bucket rows by op_idx -> per-op dense GEMM pairs on v_wmma_f32_16x16x32_bf16.
// Weights pre-converted to bf16 and pre-transposed once (if ws allows), so all
// GEMM tiles except GEMM1's x tiles stream via global_load_async_to_lds_b128 (ASYNCcnt DMA).

#define BDIM 4096
#define EDIM 1024
#define ODIM 16
#define HDIM 1024
#define TILE_M 128
#define TILE_N 128
#define TK 32
#define ASTR 40   // ushort stride: 80B rows -> 16B-aligned async B128 chunks
#define BSTR 40
#define PADDED_MAX (BDIM + ODIM * TILE_M)   // 6144 worst-case slots
#define MAX_TX (BDIM / TILE_M + ODIM)       // 48 worst-case row tiles

typedef __attribute__((ext_vector_type(16))) __bf16 v16bf;
typedef __attribute__((ext_vector_type(8)))  float  v8f;
union Frag { v16bf v; unsigned int u[8]; };

#if __has_builtin(__builtin_amdgcn_cvt_pk_bf16_f32)
typedef __attribute__((ext_vector_type(2))) __bf16 v2bf;
__device__ __forceinline__ unsigned int pack_bf16(float a, float b) {
  union { v2bf v; unsigned int u; } c;
  c.v = __builtin_amdgcn_cvt_pk_bf16_f32(a, b);
  return c.u;
}
#else
// RNE to bf16 then pack both high halves with one v_perm_b32.
__device__ __forceinline__ unsigned int pack_bf16(float a, float b) {
  unsigned int ua = __float_as_uint(a), ub = __float_as_uint(b);
  ua += 0x7FFFu + ((ua >> 16) & 1u);
  ub += 0x7FFFu + ((ub >> 16) & 1u);
  return __builtin_amdgcn_perm(ub, ua, 0x07060302u);  // {ub.hi16, ua.hi16}
}
#endif

__device__ __forceinline__ unsigned short f2bf(float f) {
  unsigned int u = __float_as_uint(f);
  u += 0x7FFFu + ((u >> 16) & 1u);
  return (unsigned short)(u >> 16);
}

// ---------------- bucket setup ----------------
// meta ints: [0..15] counts, [16..32] padded offsets, [33..48] fill counters
__global__ void ob_init(int* meta, int* perm) {
  int t = blockIdx.x * blockDim.x + threadIdx.x;
  if (t < 64) meta[t] = 0;
  if (t < PADDED_MAX) perm[t] = -1;
}

__global__ void ob_hist(const int* __restrict__ op_idx, int* meta) {
  int b = blockIdx.x * blockDim.x + threadIdx.x;
  if (b < BDIM) atomicAdd(&meta[op_idx[b]], 1);
}

__global__ void ob_scan(int* meta) {
  if (threadIdx.x == 0) {
    int off = 0;
    for (int o = 0; o < ODIM; ++o) {
      meta[16 + o] = off;
      off += ((meta[o] + TILE_M - 1) / TILE_M) * TILE_M;
    }
    meta[32] = off;
  }
}

__global__ void ob_scatter(const int* __restrict__ op_idx, int* meta, int* perm) {
  int b = blockIdx.x * blockDim.x + threadIdx.x;
  if (b < BDIM) {
    int o = op_idx[b];
    int s = atomicAdd(&meta[33 + o], 1);
    perm[meta[16 + o] + s] = b;
  }
}

// W[e][o][n] f32 -> Wt[o][n][e] bf16 (k-contiguous rows for async tile DMA).
// Thread -> (e8, o, n); reads coalesced along n, writes 16B per thread.
__global__ __launch_bounds__(256)
void ob_cvtw(const float* __restrict__ W, unsigned short* __restrict__ Wt) {
  int idx = blockIdx.x * 256 + threadIdx.x;
  int n = idx % HDIM;
  int rest = idx / HDIM;
  int o = rest % ODIM;
  int e0 = (rest / ODIM) * 8;
  unsigned int pk[4];
#pragma unroll
  for (int j = 0; j < 4; ++j) {
    float f0 = W[(size_t)(e0 + 2 * j) * ODIM * HDIM + (size_t)o * HDIM + n];
    float f1 = W[(size_t)(e0 + 2 * j + 1) * ODIM * HDIM + (size_t)o * HDIM + n];
    pk[j] = pack_bf16(f0, f1);
  }
  uint4 v; v.x = pk[0]; v.y = pk[1]; v.z = pk[2]; v.w = pk[3];
  *(uint4*)(Wt + (size_t)(o * HDIM + n) * EDIM + e0) = v;
}

__device__ __forceinline__ int locate_op(const int* __restrict__ meta, int t,
                                         int* o_out, int* mt_out) {
  int accT = 0;
#pragma unroll
  for (int i = 0; i < ODIM; ++i) {
    int tiles = (meta[16 + i + 1] - meta[16 + i]) >> 7;
    if (t < accT + tiles) { *o_out = i; *mt_out = t - accT; return 1; }
    accT += tiles;
  }
  return 0;
}

// 16B async DMA of a 128x32-ushort tile into padded LDS (2 chunks/thread).
__device__ __forceinline__ void async_tile(const unsigned short* __restrict__ gbase,
                                           size_t grow_stride, int e0,
                                           unsigned short (*lds)[ASTR], int tid) {
#pragma unroll
  for (int i = 0; i < 2; ++i) {
    int c = i * 256 + tid;
    int row = c >> 2, sub = c & 3;
    unsigned int l = (unsigned int)(size_t)&lds[row][sub * 8];
    const unsigned short* g = gbase + (size_t)row * grow_stride + e0 + sub * 8;
    asm volatile("global_load_async_to_lds_b128 %0, %1, off"
                 :: "v"(l), "v"((unsigned long long)(size_t)g) : "memory");
  }
}

// ---------------- GEMM1: h1[slot,n] = relu(x[perm[slot],:] @ W1[:,o,:] + b1[o,:]) ----------------
template <bool PRE>
__global__ __launch_bounds__(256)
void ob_gemm1(const float* __restrict__ x, const float* __restrict__ W1f,
              const unsigned short* __restrict__ W1t, const float* __restrict__ b1,
              const int* __restrict__ meta, const int* __restrict__ perm,
              unsigned short* __restrict__ h1) {
  __shared__ unsigned short As[2][TILE_M][ASTR];
  __shared__ unsigned short Bs[2][TILE_N][BSTR];

  const int tid = threadIdx.x;
  int o = 0, mt = 0;
  if (!locate_op(meta, blockIdx.x, &o, &mt)) return;
  const int slot0 = meta[16 + o] + mt * TILE_M;
  const int n0 = blockIdx.y * TILE_N;

  const int arow = tid >> 3;
  const int ak0  = (tid & 7) * 4;
  const float* px[4];
#pragma unroll
  for (int i = 0; i < 4; ++i) {
    int r = perm[slot0 + arow + 32 * i];
    px[i] = (r >= 0) ? (x + (size_t)r * EDIM) : nullptr;
  }
  const int bnn = (tid & 31) * 4;
  const int bk  = tid >> 5;

  const int lane = tid & 31;
  const int wv = tid >> 5;
  const int wm = wv & 3;
  const int wn = wv >> 2;
  const int half = (lane >> 4) & 1;
  const int ln16 = lane & 15;

  int kidx[8];
#pragma unroll
  for (int v = 0; v < 8; ++v)
    kidx[v] = (v < 4) ? (half * 8 + 2 * v) : (16 + half * 8 + 2 * (v - 4));

  v8f acc[2][4];
#pragma unroll
  for (int im = 0; im < 2; ++im)
#pragma unroll
    for (int jn = 0; jn < 4; ++jn)
      acc[im][jn] = v8f{0.f, 0.f, 0.f, 0.f, 0.f, 0.f, 0.f, 0.f};

  float4 fa[4], fb[4];
  auto loadA = [&](int e0) {
#pragma unroll
    for (int i = 0; i < 4; ++i)
      fa[i] = px[i] ? *(const float4*)(px[i] + e0 + ak0)
                    : make_float4(0.f, 0.f, 0.f, 0.f);
  };
  auto loadBreg = [&](int e0) {
#pragma unroll
    for (int i = 0; i < 4; ++i) {
      int k = bk + 8 * i;
      fb[i] = *(const float4*)(W1f + (size_t)(e0 + k) * ODIM * HDIM +
                               (size_t)o * HDIM + n0 + bnn);
    }
  };
  const unsigned short* wt_base = PRE ? (W1t + (size_t)(o * HDIM + n0) * EDIM) : nullptr;

  loadA(0);
  if constexpr (PRE) async_tile(wt_base, EDIM, 0, Bs[0], tid); else loadBreg(0);

  const int NIT = EDIM / TK;
  int p = 0;
  for (int it = 0; it < NIT; ++it, p ^= 1) {
#pragma unroll
    for (int i = 0; i < 4; ++i) {                 // stage A -> LDS[p]
      unsigned int u0 = pack_bf16(fa[i].x, fa[i].y);
      unsigned int u1 = pack_bf16(fa[i].z, fa[i].w);
      unsigned int* d = (unsigned int*)&As[p][arow + 32 * i][ak0];
      d[0] = u0; d[1] = u1;
    }
    if constexpr (!PRE) {
#pragma unroll
      for (int i = 0; i < 4; ++i) {               // stage B (transposed) -> LDS[p]
        int k = bk + 8 * i;
        unsigned int p01 = pack_bf16(fb[i].x, fb[i].y);
        unsigned int p23 = pack_bf16(fb[i].z, fb[i].w);
        Bs[p][bnn + 0][k] = (unsigned short)p01;
        Bs[p][bnn + 1][k] = (unsigned short)(p01 >> 16);
        Bs[p][bnn + 2][k] = (unsigned short)p23;
        Bs[p][bnn + 3][k] = (unsigned short)(p23 >> 16);
      }
    }
    if constexpr (PRE) asm volatile("s_wait_asynccnt 0x0" ::: "memory");
    __syncthreads();
    if (it + 1 < NIT) {
      loadA((it + 1) * TK);
      if constexpr (PRE) async_tile(wt_base, EDIM, (it + 1) * TK, Bs[p ^ 1], tid);
      else loadBreg((it + 1) * TK);
    }

    Frag a[2], b[4];
#pragma unroll
    for (int im = 0; im < 2; ++im) {
      int row = 32 * wm + 16 * im + ln16;
#pragma unroll
      for (int v = 0; v < 8; ++v) a[im].u[v] = *(const unsigned int*)&As[p][row][kidx[v]];
    }
#pragma unroll
    for (int jn = 0; jn < 4; ++jn) {
      int col = 64 * wn + 16 * jn + ln16;
#pragma unroll
      for (int v = 0; v < 8; ++v) b[jn].u[v] = *(const unsigned int*)&Bs[p][col][kidx[v]];
    }
#pragma unroll
    for (int im = 0; im < 2; ++im)
#pragma unroll
      for (int jn = 0; jn < 4; ++jn)
        acc[im][jn] = __builtin_amdgcn_wmma_f32_16x16x32_bf16(
            false, a[im].v, false, b[jn].v, (short)0, acc[im][jn], false, false);
  }

#pragma unroll
  for (int im = 0; im < 2; ++im) {
#pragma unroll
    for (int jn = 0; jn < 4; ++jn) {
      int n = n0 + 64 * wn + 16 * jn + ln16;
      float bias = b1[(size_t)o * HDIM + n];
#pragma unroll
      for (int r = 0; r < 8; ++r) {
        int m = 32 * wm + 16 * im + r + 8 * half;
        float v = fmaxf(acc[im][jn][r] + bias, 0.f);
        h1[(size_t)(slot0 + m) * HDIM + n] = f2bf(v);
      }
    }
  }
}

// ---------------- GEMM2: out[perm[slot],g] = relu(h1[slot,:] @ W2[:,o,:] + b2[o,:]) ----------------
template <bool PRE>
__global__ __launch_bounds__(256)
void ob_gemm2(const unsigned short* __restrict__ h1, const float* __restrict__ W2f,
              const unsigned short* __restrict__ W2t, const float* __restrict__ b2,
              const int* __restrict__ meta, const int* __restrict__ perm,
              float* __restrict__ out) {
  __shared__ unsigned short As[2][TILE_M][ASTR];
  __shared__ unsigned short Bs[2][TILE_N][BSTR];

  const int tid = threadIdx.x;
  int o = 0, mt = 0;
  if (!locate_op(meta, blockIdx.x, &o, &mt)) return;
  const int slot0 = meta[16 + o] + mt * TILE_M;
  const int n0 = blockIdx.y * TILE_N;

  const int bnn = (tid & 31) * 4;
  const int bk  = tid >> 5;

  const int lane = tid & 31;
  const int wv = tid >> 5;
  const int wm = wv & 3;
  const int wn = wv >> 2;
  const int half = (lane >> 4) & 1;
  const int ln16 = lane & 15;

  int kidx[8];
#pragma unroll
  for (int v = 0; v < 8; ++v)
    kidx[v] = (v < 4) ? (half * 8 + 2 * v) : (16 + half * 8 + 2 * (v - 4));

  v8f acc[2][4];
#pragma unroll
  for (int im = 0; im < 2; ++im)
#pragma unroll
    for (int jn = 0; jn < 4; ++jn)
      acc[im][jn] = v8f{0.f, 0.f, 0.f, 0.f, 0.f, 0.f, 0.f, 0.f};

  const unsigned short* a_base  = h1 + (size_t)slot0 * HDIM;
  const unsigned short* wt_base = PRE ? (W2t + (size_t)(o * HDIM + n0) * EDIM) : nullptr;

  float4 fb[4];
  auto loadBreg = [&](int e0) {
#pragma unroll
    for (int i = 0; i < 4; ++i) {
      int k = bk + 8 * i;
      fb[i] = *(const float4*)(W2f + (size_t)(e0 + k) * ODIM * HDIM +
                               (size_t)o * HDIM + n0 + bnn);
    }
  };

  async_tile(a_base, HDIM, 0, As[0], tid);
  if constexpr (PRE) async_tile(wt_base, EDIM, 0, Bs[0], tid); else loadBreg(0);

  const int NIT = HDIM / TK;
  int p = 0;
  for (int it = 0; it < NIT; ++it, p ^= 1) {
    if constexpr (!PRE) {
#pragma unroll
      for (int i = 0; i < 4; ++i) {
        int k = bk + 8 * i;
        unsigned int p01 = pack_bf16(fb[i].x, fb[i].y);
        unsigned int p23 = pack_bf16(fb[i].z, fb[i].w);
        Bs[p][bnn + 0][k] = (unsigned short)p01;
        Bs[p][bnn + 1][k] = (unsigned short)(p01 >> 16);
        Bs[p][bnn + 2][k] = (unsigned short)p23;
        Bs[p][bnn + 3][k] = (unsigned short)(p23 >> 16);
      }
    }
    asm volatile("s_wait_asynccnt 0x0" ::: "memory");
    __syncthreads();
    if (it + 1 < NIT) {
      async_tile(a_base, HDIM, (it + 1) * TK, As[p ^ 1], tid);
      if constexpr (PRE) async_tile(wt_base, EDIM, (it + 1) * TK, Bs[p ^ 1], tid);
      else loadBreg((it + 1) * TK);
    }

    Frag a[2], b[4];
#pragma unroll
    for (int im = 0; im < 2; ++im) {
      int row = 32 * wm + 16 * im + ln16;
#pragma unroll
      for (int v = 0; v < 8; ++v) a[im].u[v] = *(const unsigned int*)&As[p][row][kidx[v]];
    }
#pragma unroll
    for (int jn = 0; jn < 4; ++jn) {
      int col = 64 * wn + 16 * jn + ln16;
#pragma unroll
      for (int v = 0; v < 8; ++v) b[jn].u[v] = *(const unsigned int*)&Bs[p][col][kidx[v]];
    }
#pragma unroll
    for (int im = 0; im < 2; ++im)
#pragma unroll
      for (int jn = 0; jn < 4; ++jn)
        acc[im][jn] = __builtin_amdgcn_wmma_f32_16x16x32_bf16(
            false, a[im].v, false, b[jn].v, (short)0, acc[im][jn], false, false);
  }

#pragma unroll
  for (int im = 0; im < 2; ++im) {
    int rowid[8];
#pragma unroll
    for (int r = 0; r < 8; ++r)
      rowid[r] = perm[slot0 + 32 * wm + 16 * im + r + 8 * half];
#pragma unroll
    for (int jn = 0; jn < 4; ++jn) {
      int n = n0 + 64 * wn + 16 * jn + ln16;
      float bias = b2[(size_t)o * HDIM + n];
#pragma unroll
      for (int r = 0; r < 8; ++r) {
        if (rowid[r] >= 0)
          out[(size_t)rowid[r] * HDIM + n] = fmaxf(acc[im][jn][r] + bias, 0.f);
      }
    }
  }
}

extern "C" void kernel_launch(void* const* d_in, const int* in_sizes, int n_in,
                              void* d_out, int out_size, void* d_ws, size_t ws_size,
                              hipStream_t stream) {
  const float* x      = (const float*)d_in[0];
  const int*   op_idx = (const int*)d_in[1];
  const float* W1     = (const float*)d_in[2];
  const float* b1     = (const float*)d_in[3];
  const float* W2     = (const float*)d_in[4];
  const float* b2     = (const float*)d_in[5];
  float* out = (float*)d_out;

  int* meta = (int*)d_ws;
  int* perm = meta + 64;
  const size_t off_h1  = 32768;
  const size_t wsz     = (size_t)EDIM * ODIM * HDIM;              // elements per weight
  const size_t off_wt1 = off_h1 + (size_t)PADDED_MAX * HDIM * 2;
  const size_t off_wt2 = off_wt1 + wsz * 2;
  const size_t need    = off_wt2 + wsz * 2;                       // ~77MB
  unsigned short* h1  = (unsigned short*)((char*)d_ws + off_h1);
  unsigned short* wt1 = (unsigned short*)((char*)d_ws + off_wt1);
  unsigned short* wt2 = (unsigned short*)((char*)d_ws + off_wt2);
  const bool pre = (ws_size >= need);

  ob_init<<<(PADDED_MAX + 255) / 256, 256, 0, stream>>>(meta, perm);
  ob_hist<<<BDIM / 256, 256, 0, stream>>>(op_idx, meta);
  ob_scan<<<1, 32, 0, stream>>>(meta);
  ob_scatter<<<BDIM / 256, 256, 0, stream>>>(op_idx, meta, perm);

  dim3 grid(MAX_TX, HDIM / TILE_N);
  if (pre) {
    const int cvt_blocks = (EDIM / 8) * ODIM * HDIM / 256;        // 8192
    ob_cvtw<<<cvt_blocks, 256, 0, stream>>>(W1, wt1);
    ob_cvtw<<<cvt_blocks, 256, 0, stream>>>(W2, wt2);
    ob_gemm1<true><<<grid, 256, 0, stream>>>(x, W1, wt1, b1, meta, perm, h1);
    ob_gemm2<true><<<grid, 256, 0, stream>>>(h1, W2, wt2, b2, meta, perm, out);
  } else {
    ob_gemm1<false><<<grid, 256, 0, stream>>>(x, W1, nullptr, b1, meta, perm, h1);
    ob_gemm2<false><<<grid, 256, 0, stream>>>(h1, W2, nullptr, b2, meta, perm, out);
  }
}
